// LSTM_747324310249
// MI455X (gfx1250) — matrix operand order repaired and checked by reading, not compiled, not run on hardware
//
#include <hip/hip_runtime.h>
#include <cstdint>

typedef __bf16 bf16_t;
typedef __attribute__((ext_vector_type(16))) __bf16 v16bf;
typedef __attribute__((ext_vector_type(8)))  __bf16 v8bf;
typedef __attribute__((ext_vector_type(8)))  float  v8f;
typedef unsigned int u32x4 __attribute__((ext_vector_type(4)));
typedef int          i32x4 __attribute__((ext_vector_type(4)));
typedef int          i32x8 __attribute__((ext_vector_type(8)));

constexpr int S  = 256;
constexpr int B  = 64;
constexpr int IN = 512;
constexpr int H  = 1024;
constexpr int NG = 4 * H;   // 4096 gate columns (f,i,o,c)

#if defined(__HIP_DEVICE_COMPILE__)
#if __has_builtin(__builtin_amdgcn_tensor_load_to_lds) && \
    __has_builtin(__builtin_amdgcn_s_wait_tensorcnt)
#define USE_TDM 1
#endif
#endif

union AFrag { v16bf v; v8bf h[2]; bf16_t e[16]; };

// A-fragment (16x32, M x K) from bf16 row-major [row][k], leading dim ldk.
// Lanes 0-15: K = {k0..k0+7, k0+16..k0+23}; lanes 16-31: +8.
__device__ __forceinline__ v16bf load_a_bf16(const bf16_t* base, int ldk,
                                             int row, int k0, int half) {
    const bf16_t* p = base + (size_t)row * ldk + k0 + 8 * half;
    AFrag f;
    f.h[0] = *(const v8bf*)p;
    f.h[1] = *(const v8bf*)(p + 16);
    return f.v;
}

// A-fragment from f32 source (convert to bf16 on the fly).
__device__ __forceinline__ v16bf load_a_f32(const float* __restrict__ base,
                                            int ldk, int row, int k0, int half) {
    const float* p = base + (size_t)row * ldk + k0 + 8 * half;
    float4 f0 = *(const float4*)(p);
    float4 f1 = *(const float4*)(p + 4);
    float4 f2 = *(const float4*)(p + 16);
    float4 f3 = *(const float4*)(p + 20);
    AFrag f;
    f.e[0]  = (bf16_t)f0.x; f.e[1]  = (bf16_t)f0.y; f.e[2]  = (bf16_t)f0.z; f.e[3]  = (bf16_t)f0.w;
    f.e[4]  = (bf16_t)f1.x; f.e[5]  = (bf16_t)f1.y; f.e[6]  = (bf16_t)f1.z; f.e[7]  = (bf16_t)f1.w;
    f.e[8]  = (bf16_t)f2.x; f.e[9]  = (bf16_t)f2.y; f.e[10] = (bf16_t)f2.z; f.e[11] = (bf16_t)f2.w;
    f.e[12] = (bf16_t)f3.x; f.e[13] = (bf16_t)f3.y; f.e[14] = (bf16_t)f3.z; f.e[15] = (bf16_t)f3.w;
    return f.v;
}

// B-fragment (32x16, K x N) where B[k][n] = W[n][k], W bf16 row-major [n][k].
// Lane (half,r): column n=r, K = {k0+8*half+(0..7), k0+16+8*half+(0..7)}.
__device__ __forceinline__ v16bf load_b_bf16(const bf16_t* __restrict__ base,
                                             int ldk, int n, int k0, int half) {
    const bf16_t* p = base + (size_t)n * ldk + k0 + 8 * half;
    AFrag f;
    f.h[0] = *(const v8bf*)p;
    f.h[1] = *(const v8bf*)(p + 16);
    return f.v;
}

#if defined(USE_TDM)
// One-shot 1-D TDM DMA: ndw dwords from global memory into LDS at lds_off.
// D# per CDNA5 ISA ch.8: group0 {count=1, lds_addr, global_addr, type=2},
// group1 {data_size=4B, tensor_dim0=ndw, tile_dim0=ndw}; groups 2/3 zero (<=2D).
__device__ __forceinline__ void tdm_copy_to_lds(uint32_t lds_off, const void* gptr,
                                                uint32_t ndw) {
    uint64_t ga = (uint64_t)(uintptr_t)gptr;
    u32x4 g0;
    g0[0] = 1u;                                                  // count=1 (valid user D#)
    g0[1] = lds_off;                                             // lds_addr (bytes)
    g0[2] = (uint32_t)ga;                                        // global_addr[31:0]
    g0[3] = (uint32_t)((ga >> 32) & 0x1FFFFFFu) | (2u << 30);    // addr[56:32] | type=2
    i32x8 g1;
    g1[0] = (int)(2u << 16);                                     // data_size=2 (4 bytes)
    g1[1] = (int)((ndw & 0xFFFFu) << 16);                        // tensor_dim0[15:0]
    g1[2] = (int)((ndw >> 16) & 0xFFFFu) | (1 << 16);            // dim0[31:16], tensor_dim1=1
    g1[3] = (int)((ndw & 0xFFFFu) << 16);                        // tile_dim0 = ndw
    g1[4] = 0;                                                   // tile_dim1/2 unused
    g1[5] = (int)ndw;                                            // tensor_dim0_stride lo
    g1[6] = 0;
    g1[7] = 0;
    i32x4 z4 = {0, 0, 0, 0};
#if __clang_major__ >= 23
    i32x8 z8 = {0, 0, 0, 0, 0, 0, 0, 0};
    __builtin_amdgcn_tensor_load_to_lds(g0, g1, z4, z4, z8, 0);
#else
    __builtin_amdgcn_tensor_load_to_lds(g0, g1, z4, z4, 0);
#endif
}
#endif

// ---------------- packing / init kernels ----------------

__global__ void pack_w(const float* __restrict__ w0, const float* __restrict__ w1,
                       const float* __restrict__ w2, const float* __restrict__ w3,
                       bf16_t* __restrict__ dst, int log2pg) {
    size_t idx = (size_t)blockIdx.x * blockDim.x + threadIdx.x;
    int g = (int)(idx >> log2pg);
    size_t off = idx & (((size_t)1 << log2pg) - 1);
    const float* src = (g == 0) ? w0 : (g == 1) ? w1 : (g == 2) ? w2 : w3;
    dst[idx] = (bf16_t)src[off];
}

__global__ void pack_bias(const float* __restrict__ bfx, const float* __restrict__ bix,
                          const float* __restrict__ box, const float* __restrict__ bcx,
                          const float* __restrict__ bfh, const float* __restrict__ bih,
                          const float* __restrict__ boh, const float* __restrict__ bch,
                          float* __restrict__ bias) {
    int i = blockIdx.x * blockDim.x + threadIdx.x;   // [0, 4096)
    int g = i >> 10, j = i & (H - 1);
    const float* bx = (g == 0) ? bfx : (g == 1) ? bix : (g == 2) ? box : bcx;
    const float* bh = (g == 0) ? bfh : (g == 1) ? bih : (g == 2) ? boh : bch;
    bias[i] = bx[j] + bh[j];
}

__global__ void init_state(bf16_t* __restrict__ h0, float* __restrict__ c) {
    int i = blockIdx.x * blockDim.x + threadIdx.x;   // [0, B*H)
    h0[i] = (bf16_t)0.0f;
    c[i]  = 0.0f;
}

// ---------------- gx = x @ Wx^T  (M=S*B=16384, N=4096, K=512) ----------------
// WG: 256 threads (8 waves). WG tile: 128(M) x 64(N). Wave: 16(M) x 64(N).

__global__ __launch_bounds__(256) void gx_gemm(const float* __restrict__ x,
                                               const bf16_t* __restrict__ Wxb,
                                               bf16_t* __restrict__ gxb) {
    const int lane = threadIdx.x & 31;
    const int wave = threadIdx.x >> 5;
    const int half = lane >> 4, r = lane & 15;
    const int n0 = blockIdx.x * 64;
    const int m0 = blockIdx.y * 128 + wave * 16;

    v8f acc[4] = {};
    for (int k0 = 0; k0 < IN; k0 += 32) {
        v16bf a = load_a_f32(x, IN, m0 + r, k0, half);
#pragma unroll
        for (int nt = 0; nt < 4; ++nt) {
            v16bf b = load_b_bf16(Wxb, IN, n0 + nt * 16 + r, k0, half);
            acc[nt] = __builtin_amdgcn_wmma_f32_16x16x32_bf16(
                false, a, false, b, (short)0, acc[nt], false, false);
        }
    }
#pragma unroll
    for (int nt = 0; nt < 4; ++nt)
#pragma unroll
        for (int v = 0; v < 8; ++v)
            gxb[(size_t)(m0 + v + 8 * half) * NG + n0 + nt * 16 + r] = (bf16_t)acc[nt][v];
}

// ---------------- recurrent step: gh = h @ Wh^T, gates, cell update --------
// Grid: H/16 = 64 WGs; WG handles 16 hidden columns (j0..j0+15) of all 4 gates.
// h_prev (128 KB bf16) is DMA'd into LDS once per WG by the Tensor Data Mover,
// then A-fragments come from LDS (kills the 4x redundant L2 A-traffic).
// Wave w: gate g = w>>1, batch half m0 = (w&1)*32 -> two 16x16 tiles, K=1024.

__global__ __launch_bounds__(256) void lstm_step(const bf16_t* __restrict__ hprev,
                                                 bf16_t* __restrict__ hnext,
                                                 const bf16_t* __restrict__ Whb,
                                                 const float* __restrict__ bias,
                                                 const bf16_t* __restrict__ gxb,
                                                 float* __restrict__ c_ws,
                                                 float* __restrict__ h_seq, int t) {
    __shared__ float gh[4][B][16];
    extern __shared__ char smem_dyn[];                 // B*H bf16 = 128 KB
    bf16_t* hs = (bf16_t*)smem_dyn;

#if defined(USE_TDM)
    if (threadIdx.x < 32) {                            // wave 0 issues one DMA
        tdm_copy_to_lds((uint32_t)(uintptr_t)hs, hprev,
                        (uint32_t)(B * H * sizeof(bf16_t) / 4));
        __builtin_amdgcn_s_wait_tensorcnt(0);
    }
#else
    for (int i = threadIdx.x; i < (B * H) / 8; i += 256)
        ((v8bf*)hs)[i] = ((const v8bf*)hprev)[i];
#endif
    __syncthreads();

    const int lane = threadIdx.x & 31;
    const int wave = threadIdx.x >> 5;
    const int half = lane >> 4, r = lane & 15;
    const int g  = wave >> 1;
    const int m0 = (wave & 1) * 32;
    const int j0 = blockIdx.x * 16;
    const int n  = g * H + j0 + r;       // weight row for this lane's column

    v8f acc0 = {}, acc1 = {};
    for (int k0 = 0; k0 < H; k0 += 32) {
        if (k0 + 64 < H)                 // speculative prefetch of next B chunk
            __builtin_prefetch(Whb + (size_t)n * H + k0 + 64, 0, 1);
        v16bf b  = load_b_bf16(Whb, H, n, k0, half);
        v16bf a0 = load_a_bf16(hs, H, m0 + r, k0, half);
        v16bf a1 = load_a_bf16(hs, H, m0 + 16 + r, k0, half);
        acc0 = __builtin_amdgcn_wmma_f32_16x16x32_bf16(false, a0, false, b, (short)0, acc0, false, false);
        acc1 = __builtin_amdgcn_wmma_f32_16x16x32_bf16(false, a1, false, b, (short)0, acc1, false, false);
    }
#pragma unroll
    for (int v = 0; v < 8; ++v) {
        gh[g][m0 + v + 8 * half][r]      = acc0[v];
        gh[g][m0 + 16 + v + 8 * half][r] = acc1[v];
    }
    __syncthreads();

    const bf16_t* gx_t = gxb + (size_t)t * B * NG;
#pragma unroll
    for (int i = 0; i < 4; ++i) {
        int e   = threadIdx.x + i * 256;     // 64 rows x 16 cols = 1024 elems
        int row = e >> 4, col = e & 15;
        int j   = j0 + col;
        float gf = gh[0][row][col] + (float)gx_t[(size_t)row * NG + 0 * H + j] + bias[0 * H + j];
        float gi = gh[1][row][col] + (float)gx_t[(size_t)row * NG + 1 * H + j] + bias[1 * H + j];
        float go = gh[2][row][col] + (float)gx_t[(size_t)row * NG + 2 * H + j] + bias[2 * H + j];
        float gc = gh[3][row][col] + (float)gx_t[(size_t)row * NG + 3 * H + j] + bias[3 * H + j];
        float fg = 1.0f / (1.0f + __expf(-gf));
        float ig = 1.0f / (1.0f + __expf(-gi));
        float og = 1.0f / (1.0f + __expf(-go));
        float cold = c_ws[row * H + j];
        float cnew = fg * cold + ig * tanhf(gc);
        float hnew = og * tanhf(cnew);
        c_ws[row * H + j] = cnew;
        h_seq[((size_t)t * B + row) * H + j] = hnew;
        hnext[row * H + j] = (bf16_t)hnew;
    }
}

__global__ void finalize(const float* __restrict__ h_last, const float* __restrict__ c_ws,
                         float* __restrict__ out_h, float* __restrict__ out_c) {
    int i = blockIdx.x * blockDim.x + threadIdx.x;   // [0, B*H)
    out_h[i] = h_last[i];
    out_c[i] = c_ws[i];
}

// ---------------- launcher ----------------

extern "C" void kernel_launch(void* const* d_in, const int* in_sizes, int n_in,
                              void* d_out, int out_size, void* d_ws, size_t ws_size,
                              hipStream_t stream) {
    (void)in_sizes; (void)n_in; (void)out_size; (void)ws_size;

    const float* x   = (const float*)d_in[0];
    const float* Wfx = (const float*)d_in[1];  const float* bfx = (const float*)d_in[2];
    const float* Wix = (const float*)d_in[3];  const float* bix = (const float*)d_in[4];
    const float* Wox = (const float*)d_in[5];  const float* box = (const float*)d_in[6];
    const float* Wcx = (const float*)d_in[7];  const float* bcx = (const float*)d_in[8];
    const float* Wfh = (const float*)d_in[9];  const float* bfh = (const float*)d_in[10];
    const float* Wih = (const float*)d_in[11]; const float* bih = (const float*)d_in[12];
    const float* Woh = (const float*)d_in[13]; const float* boh = (const float*)d_in[14];
    const float* Wch = (const float*)d_in[15]; const float* bch = (const float*)d_in[16];

    char* ws = (char*)d_ws;
    bf16_t* gxb  = (bf16_t*)ws; ws += (size_t)S * B * NG * sizeof(bf16_t);  // 128 MB
    bf16_t* Wxb  = (bf16_t*)ws; ws += (size_t)NG * IN * sizeof(bf16_t);     //   4 MB
    bf16_t* Whb  = (bf16_t*)ws; ws += (size_t)NG * H * sizeof(bf16_t);      //   8 MB
    float*  bias = (float*) ws; ws += (size_t)NG * sizeof(float);
    bf16_t* hb0  = (bf16_t*)ws; ws += (size_t)B * H * sizeof(bf16_t);
    bf16_t* hb1  = (bf16_t*)ws; ws += (size_t)B * H * sizeof(bf16_t);
    float*  c_ws = (float*) ws; ws += (size_t)B * H * sizeof(float);

    pack_w<<<(4 * H * IN) / 256, 256, 0, stream>>>(Wfx, Wix, Wox, Wcx, Wxb, 19);
    pack_w<<<(4 * H * H) / 256, 256, 0, stream>>>(Wfh, Wih, Woh, Wch, Whb, 20);
    pack_bias<<<NG / 256, 256, 0, stream>>>(bfx, bix, box, bcx, bfh, bih, boh, bch, bias);
    init_state<<<(B * H) / 256, 256, 0, stream>>>(hb0, c_ws);

    gx_gemm<<<dim3(NG / 64, (S * B) / 128), 256, 0, stream>>>(x, Wxb, gxb);

    float* h_seq = (float*)d_out;
    const size_t step_lds = (size_t)B * H * sizeof(bf16_t);   // 128 KB dynamic LDS
    bf16_t* hb[2] = {hb0, hb1};
    for (int t = 0; t < S; ++t)
        lstm_step<<<H / 16, 256, step_lds, stream>>>(hb[t & 1], hb[(t + 1) & 1],
                                                     Whb, bias, gxb, c_ws, h_seq, t);

    finalize<<<(B * H) / 256, 256, 0, stream>>>(h_seq + (size_t)(S - 1) * B * H, c_ws,
                                                h_seq + (size_t)S * B * H,
                                                h_seq + (size_t)S * B * H + B * H);
}
